// ScaledDotProductAttention_11132555231519
// MI455X (gfx1250) — matrix operand order repaired and checked by reading, not compiled, not run on hardware
//
#include <hip/hip_runtime.h>
#include <hip/hip_bf16.h>
#include <math.h>

typedef __attribute__((ext_vector_type(16))) _Float16 v16h;
typedef __attribute__((ext_vector_type(8)))  float    v8f;
typedef __attribute__((ext_vector_type(4)))  float    v4f;
typedef __attribute__((ext_vector_type(4)))  int      v4i;

#define ATTN_S   2048
#define ATTN_D   64
#define QTILE    16
#define NTHREADS 256
#define NWAVES   8
#define KPW      (ATTN_S / NWAVES)   // 256 k-columns per wave
#define SROW     (ATTN_S + 4)        // padded LDS row stride: 2052 % 64banks = 4,
                                     // 16B-aligned rows for float4 LDS access

extern __shared__ char smem_raw[];

__launch_bounds__(NTHREADS, 2)
__global__ void sdpa_fused_kernel(const float* __restrict__ Q,
                                  const float* __restrict__ Kin,
                                  const float* __restrict__ Vin,
                                  const int*   __restrict__ Mask,
                                  float* __restrict__ ctx_out,
                                  float* __restrict__ attn_out)
{
    float* sS   = (float*)smem_raw;            // [QTILE][SROW]  scores -> probs
    float* sC   = sS  + QTILE * SROW;          // [QTILE][ATTN_D] context accumulator
    float* sRed = sC  + QTILE * ATTN_D;        // [256] reduction scratch
    float* sMax = sRed + NTHREADS;             // [16]
    float* sSum = sMax + QTILE;                // [16]

    const int tid    = threadIdx.x;
    const int wave   = tid >> 5;
    const int lane   = tid & 31;
    const int laneLo = lane & 15;
    const int laneHi = lane >> 4;

    const int bid   = blockIdx.x;
    const int bh    = bid >> 7;                // 128 q-tiles per (b,h)
    const int qt    = bid & 127;
    const int qbase = qt * QTILE;

    const float* Qp = Q   + ((size_t)bh * ATTN_S + qbase) * ATTN_D;
    const float* Kp = Kin + (size_t)bh * ATTN_S * ATTN_D;
    const float* Vp = Vin + (size_t)bh * ATTN_S * ATTN_D;
    const int*   Mp = Mask + ((size_t)bh * ATTN_S + qbase) * (size_t)ATTN_S;
    float* attn_o = attn_out + ((size_t)bh * ATTN_S + qbase) * (size_t)ATTN_S;
    float* ctx_o  = ctx_out  + ((size_t)bh * ATTN_S + qbase) * ATTN_D;

    // zero context accumulator tile (visible after the first barrier)
    for (int i = tid; i < QTILE * ATTN_D; i += NTHREADS) sC[i] = 0.0f;

    // ---- Q A-fragments (f32 -> f16), reused across every k-tile ----
    // A 16x32 layout: M = lane&15 ; element h -> K = h + 16*(lane>>4)
    v16h aQ0, aQ1;
#pragma unroll
    for (int h = 0; h < 16; ++h) {
        int d = h + (laneHi << 4);
        aQ0[h] = (_Float16)Qp[laneLo * ATTN_D + d];
        aQ1[h] = (_Float16)Qp[laneLo * ATTN_D + 32 + d];
    }

    const int k0 = wave * KPW;
    const int k1 = k0 + KPW;

    // ---- scores: S = (Q K^T) * 1/sqrt(64) into LDS (mask applied later) ----
    for (int kt = k0; kt < k1; kt += 16) {
        if (kt + 16 < k1)
            __builtin_prefetch(&Kp[(size_t)(kt + 16 + laneLo) * ATTN_D + laneHi * 32], 0, 1);

        // B 32x16 layout: N = lane&15 ; element h -> K = h + 16*(lane>>4)
        // B[d][k] = K[k][d]  (per lane: 16 contiguous floats of one K row)
        v16h bK0, bK1;
#pragma unroll
        for (int h = 0; h < 16; ++h) {
            int d = h + (laneHi << 4);
            bK0[h] = (_Float16)Kp[(size_t)(kt + laneLo) * ATTN_D + d];
            bK1[h] = (_Float16)Kp[(size_t)(kt + laneLo) * ATTN_D + 32 + d];
        }
        v8f acc = {};
        acc = __builtin_amdgcn_wmma_f32_16x16x32_f16(false, aQ0, false, bK0,
                                                     (short)0, acc, false, false);
        acc = __builtin_amdgcn_wmma_f32_16x16x32_f16(false, aQ1, false, bK1,
                                                     (short)0, acc, false, false);
        // D layout: element r -> (M = r + 8*laneHi, N = laneLo)
#pragma unroll
        for (int r = 0; r < 8; ++r) {
            int q = r + (laneHi << 3);
            sS[q * SROW + kt + laneLo] = acc[r] * 0.125f;
        }
    }
    __syncthreads();

    // ---- mask + row-wise softmax (16 threads per row, float4/int4 vectorized) ----
    {
        const int row = tid >> 4;
        const int sub = tid & 15;
        float*        rowS = sS + row * SROW;
        const int*    rowM = Mp + (size_t)row * ATTN_S;
        float*        rowA = attn_o + (size_t)row * ATTN_S;

        // apply mask (streaming NT load: 16 threads cover 256 contiguous bytes)
        float mx = -3.4e38f;
        for (int c = sub * 4; c < ATTN_S; c += 64) {
            v4i m = __builtin_nontemporal_load((const v4i*)(rowM + c));
            v4f s = *(v4f*)(rowS + c);
#pragma unroll
            for (int j = 0; j < 4; ++j) {
                s[j] = m[j] ? s[j] : -1.0e9f;
                mx = fmaxf(mx, s[j]);
            }
            *(v4f*)(rowS + c) = s;
        }
        sRed[row * 16 + sub] = mx;
        __syncthreads();
        if (sub == 0) {
            float m2 = sRed[row * 16];
            for (int j = 1; j < 16; ++j) m2 = fmaxf(m2, sRed[row * 16 + j]);
            sMax[row] = m2;
        }
        __syncthreads();

        const float rm = sMax[row];
        float sum = 0.0f;
        for (int c = sub * 4; c < ATTN_S; c += 64) {
            v4f s = *(v4f*)(rowS + c);
#pragma unroll
            for (int j = 0; j < 4; ++j) {
                s[j] = __expf(s[j] - rm);
                sum += s[j];
            }
            *(v4f*)(rowS + c) = s;
        }
        sRed[row * 16 + sub] = sum;
        __syncthreads();
        if (sub == 0) {
            float t = 0.0f;
            for (int j = 0; j < 16; ++j) t += sRed[row * 16 + j];
            sSum[row] = t;
        }
        __syncthreads();

        const float inv = 1.0f / sSum[row];
        for (int c = sub * 4; c < ATTN_S; c += 64) {
            v4f s = *(v4f*)(rowS + c);
#pragma unroll
            for (int j = 0; j < 4; ++j) s[j] *= inv;
            *(v4f*)(rowS + c) = s;                              // keep probs for PV
            __builtin_nontemporal_store(s, (v4f*)(rowA + c));   // 1.07 GB NT stream
        }
    }
    __syncthreads();

    // ---- context: C(16x64) += P(16x32) @ V(32x64), per-wave k-strip ----
    v8f accC[4] = {};
    for (int kc = k0; kc < k1; kc += 32) {
        v16h aP;
#pragma unroll
        for (int h = 0; h < 16; ++h) {
            int k = kc + h + (laneHi << 4);
            aP[h] = (_Float16)sS[laneLo * SROW + k];
        }
#pragma unroll
        for (int dt = 0; dt < 4; ++dt) {
            v16h bV;  // B[k][d]: per h, half-wave reads 64 contiguous bytes of V
#pragma unroll
            for (int h = 0; h < 16; ++h) {
                int kr = kc + h + (laneHi << 4);
                bV[h] = (_Float16)Vp[(size_t)kr * ATTN_D + dt * 16 + laneLo];
            }
            accC[dt] = __builtin_amdgcn_wmma_f32_16x16x32_f16(false, aP, false, bV,
                                                              (short)0, accC[dt],
                                                              false, false);
        }
    }
    // reduce the 8 per-wave partials via LDS float atomics (ds_add_f32)
#pragma unroll
    for (int dt = 0; dt < 4; ++dt)
#pragma unroll
        for (int r = 0; r < 8; ++r) {
            int q = r + (laneHi << 3);
            int d = dt * 16 + laneLo;
            atomicAdd(&sC[q * ATTN_D + d], accC[dt][r]);
        }
    __syncthreads();

    // context tile is contiguous in the output: one float4 per thread
    {
        v4f* dst = (v4f*)ctx_o;
        v4f* src = (v4f*)sC;
        dst[tid] = src[tid];
    }
}

extern "C" void kernel_launch(void* const* d_in, const int* in_sizes, int n_in,
                              void* d_out, int out_size, void* d_ws, size_t ws_size,
                              hipStream_t stream) {
    const float* Q    = (const float*)d_in[0];
    const float* K    = (const float*)d_in[1];
    const float* V    = (const float*)d_in[2];
    const int*   mask = (const int*)d_in[3];

    float* out  = (float*)d_out;
    const size_t BH = 4 * 16;
    float* ctx  = out;                                         // [B,H,S,64] first
    float* attn = out + BH * (size_t)ATTN_S * ATTN_D;          // then [B,H,S,S]

    const size_t smem_floats = (size_t)QTILE * SROW   // score/prob tile
                             + (size_t)QTILE * ATTN_D // context accumulator
                             + NTHREADS + QTILE + QTILE;
    const size_t smem = smem_floats * sizeof(float);  // ~136 KB dynamic LDS

    (void)hipFuncSetAttribute(reinterpret_cast<const void*>(sdpa_fused_kernel),
                              hipFuncAttributeMaxDynamicSharedMemorySize,
                              (int)smem);

    dim3 grid((unsigned)(BH * (ATTN_S / QTILE)));     // 8192 blocks
    sdpa_fused_kernel<<<grid, NTHREADS, smem, stream>>>(Q, K, V, mask, ctx, attn);
}
